// BilstmCRFModel_60541859004535
// MI455X (gfx1250) — matrix operand order repaired and checked by reading, not compiled, not run on hardware
//
#include <hip/hip_runtime.h>
#include <hip/hip_bf16.h>

// ---------------------------------------------------------------------------
// BiLSTM-CRF for MI455X (gfx1250, wave32).
// All GEMMs (input projection, LSTM recurrence, emit) run through
// v_wmma_f32_16x16x32_bf16 (bf16 operands, fp32 accumulate). LSTM recurrence
// is a persistent kernel: 2 directions x 16 blocks, grid barrier per step.
// Per step, h_prev (32 KB bf16) is staged global->LDS with one coalesced
// clause (1 L2 round trip instead of 16 serialized ones), WMMA A-fragments
// are then fed from LDS; all 32 Whh B-fragments stay register-resident
// (extended VGPR file) across the whole 512-step loop. c-state lives in
// registers, next-step xg slab is prefetched (global_prefetch_b8).
// ---------------------------------------------------------------------------

#define CV 50000
#define CE 256
#define CH 512
#define CB 32
#define CT 512
#define CL 48
#define CG (4*CH)      // 2048 gate width
#define CBT (CB*CT)    // 16384 rows
#define BOS 1
#define EOS 2

#define NBH 16                 // blocks per direction in LSTM kernel
#define HS  (CH/NBH)           // 32 h-columns per block
#define LSTM_BLOCKS (2*NBH)    // 32 total
#define KT  (CH/32)            // 16 k-fragments in the recurrence

typedef __attribute__((ext_vector_type(16))) __bf16 v16bf;
typedef __attribute__((ext_vector_type(8)))  float  v8f;
typedef __attribute__((ext_vector_type(8)))  unsigned short v8us;

union FragU { v16bf bf; v8us us[2]; };

__device__ __forceinline__ unsigned short f2bf(float f) {
  unsigned u = __float_as_uint(f);
  u += 0x7FFFu + ((u >> 16) & 1u);           // round-to-nearest-even
  return (unsigned short)(u >> 16);
}

// A fragment: 16x32 bf16 tile from row-major [M,K] source.
// lane r=lane&15 -> row; half h -> K in {8h..8h+7} U {16+8h..16+8h+7}.
__device__ __forceinline__ v16bf load_fragA(const unsigned short* base, int ld, int r, int h) {
  FragU f;
  const unsigned short* p = base + (size_t)r * ld + 8 * h;
  f.us[0] = *(const v8us*)(p);
  f.us[1] = *(const v8us*)(p + 16);
  return f.bf;
}

// B fragment: 32x16 bf16 tile; source stored [N,K] row-major (i.e. B^T).
// lane r -> column N; half h -> K in {8h..8h+7} U {16+8h..16+8h+7}.
__device__ __forceinline__ v16bf load_fragB(const unsigned short* base, int ld, int r, int h) {
  FragU f;
  const unsigned short* p = base + (size_t)r * ld + 8 * h;
  f.us[0] = *(const v8us*)(p);
  f.us[1] = *(const v8us*)(p + 16);
  return f.bf;
}

__device__ __forceinline__ float sigf(float x) { return 1.0f / (1.0f + __expf(-x)); }

// ---------------------------------------------------------------------------
// fp32 -> bf16 bulk convert
__global__ void f2bf_kernel(const float* __restrict__ src, unsigned short* __restrict__ dst, int n) {
  int i = blockIdx.x * blockDim.x + threadIdx.x;
  if (i < n) dst[i] = f2bf(src[i]);
}

// Embedding gather -> bf16 x [B*T, E]
__global__ void embed_kernel(const int* __restrict__ src, const float* __restrict__ emb,
                             unsigned short* __restrict__ xbf) {
  int row = blockIdx.x;            // b*T + t
  int e   = threadIdx.x;           // 0..255
  int v   = src[row];
  xbf[(size_t)row * CE + e] = f2bf(emb[(size_t)v * CE + e]);
}

// ---------------------------------------------------------------------------
// WMMA GEMM: C[M,N] = A[M,K](bf16,row-major) * Bm[N,K]^T (bf16) + bias[N].
// One wave per block computing a 16 x (NT*16) strip: the A fragment is loaded
// once per k-step and reused across NT WMMAs (NT=4 -> 4x less A traffic).
// grid = (M/16, N/(NT*16)).
template <int NT>
__global__ __launch_bounds__(32) void wmma_gemm_bias(
    const unsigned short* __restrict__ A, int lda,
    const unsigned short* __restrict__ Bm, int ldb,
    const float* __restrict__ bias,
    float* __restrict__ C, int ldc, int K) {
  int tile_m = blockIdx.x;
  int nbase  = blockIdx.y * NT;
  int lane = threadIdx.x & 31;
  int r = lane & 15, h = lane >> 4;
  const unsigned short* Abase = A + (size_t)(tile_m * 16) * lda;
  v8f acc[NT] = {};
  for (int k = 0; k < K; k += 32) {
    v16bf va = load_fragA(Abase + k, lda, r, h);
#pragma unroll
    for (int j = 0; j < NT; ++j) {
      const unsigned short* Bbase = Bm + (size_t)((nbase + j) * 16) * ldb + k;
      v16bf vb = load_fragB(Bbase, ldb, r, h);
      acc[j] = __builtin_amdgcn_wmma_f32_16x16x32_bf16(false, va, false, vb,
                                                       (short)0, acc[j], false, false);
    }
  }
#pragma unroll
  for (int j = 0; j < NT; ++j) {
    int col = (nbase + j) * 16 + r;
    float bv = bias[col];
#pragma unroll
    for (int i = 0; i < 8; ++i)
      C[(size_t)(tile_m * 16 + i + 8 * h) * ldc + col] = acc[j][i] + bv;
  }
}

// ---------------------------------------------------------------------------
// Grid-wide barrier (all LSTM blocks co-resident: 32 blocks x 256 thr)
__device__ __forceinline__ void grid_barrier(unsigned* ctr, unsigned* gen, unsigned nblocks) {
  __threadfence();               // make this thread's global stores visible (agent)
  __syncthreads();
  if (threadIdx.x == 0) {
    unsigned g = __hip_atomic_load(gen, __ATOMIC_ACQUIRE, __HIP_MEMORY_SCOPE_AGENT);
    unsigned v = __hip_atomic_fetch_add(ctr, 1u, __ATOMIC_ACQ_REL, __HIP_MEMORY_SCOPE_AGENT);
    if (v == nblocks - 1u) {
      __hip_atomic_store(ctr, 0u, __ATOMIC_RELAXED, __HIP_MEMORY_SCOPE_AGENT);
      __hip_atomic_fetch_add(gen, 1u, __ATOMIC_ACQ_REL, __HIP_MEMORY_SCOPE_AGENT);
    } else {
      while (__hip_atomic_load(gen, __ATOMIC_ACQUIRE, __HIP_MEMORY_SCOPE_AGENT) == g) {
        __builtin_amdgcn_s_sleep(2);
      }
    }
  }
  __syncthreads();
  __builtin_amdgcn_fence(__ATOMIC_ACQUIRE, "agent");  // invalidate stale lines
}

// ---------------------------------------------------------------------------
// Persistent bidirectional LSTM. grid = dim3(NBH, 2), block = 256 (8 waves).
// Block (hb, dir) owns h-columns [hb*HS, (hb+1)*HS) of its direction.
__global__ __launch_bounds__(256) void lstm_kernel(
    const float* __restrict__ xg_f, const float* __restrict__ xg_b,
    const unsigned short* __restrict__ whh_f, const unsigned short* __restrict__ whh_b,
    unsigned short* __restrict__ hcur_f, unsigned short* __restrict__ hcur_b,
    unsigned short* __restrict__ hcat, unsigned* __restrict__ hdr) {
  const int hb  = blockIdx.x;
  const int dir = blockIdx.y;
  const int hbase = hb * HS;
  const float*          xg   = dir ? xg_b  : xg_f;
  const unsigned short* whh  = dir ? whh_b : whh_f;
  unsigned short*       hcur = dir ? hcur_b : hcur_f;
  unsigned* ctr = hdr;
  unsigned* gen = hdr + 1;

  const int tid  = threadIdx.x;
  const int wave = tid >> 5;
  const int lane = tid & 31;
  const int r = lane & 15, h = lane >> 4;

  // wave -> (batch tile, gate-column tile pair); both tiles share the A frags
  const int mt    = wave >> 2;          // 0..1
  const int group = wave & 3;           // gate group i/f/g/o
  const int gcol0 = group * CH + hbase; // 16-wide tile pair at gcol0, gcol0+16

  __shared__ unsigned short h_lds[CB * CH];  // 32 KB staged h_prev (bf16)
  __shared__ float g_lds[CB][4 * HS];        // 32 x 128 f32 gate slab (16 KB)

  float creg[4] = {0.f, 0.f, 0.f, 0.f}; // persistent cell state (4 (b,lh) pairs/thread)

  // Preload this wave's Whh fragments once: resident across all 512 steps
  // (32 frags x 8 VGPRs = 256 VGPRs in the extended file).
  v16bf wb0[KT], wb1[KT];
  {
    const unsigned short* B0 = whh + (size_t)gcol0 * CH;
    const unsigned short* B1 = whh + (size_t)(gcol0 + 16) * CH;
#pragma unroll
    for (int kk = 0; kk < KT; ++kk) {
      wb0[kk] = load_fragB(B0 + kk * 32, CH, r, h);
      wb1[kk] = load_fragB(B1 + kk * 32, CH, r, h);
    }
  }

  for (int step = 0; step < CT; ++step) {
    const int t = dir ? (CT - 1 - step) : step;

    // prefetch next step's 16 KB xg slab (128 cachelines) into caches
    if (step + 1 < CT && tid < 128) {
      const int tn = dir ? (t - 1) : (t + 1);
      const int pb = tid >> 2, pg = tid & 3;
      __builtin_prefetch(&xg[((size_t)pb * CT + tn) * CG + pg * CH + hbase], 0, 3);
    }

    // -------- stage h_prev global -> LDS: one coalesced clause, 8 b128/thread
    // (all loads in flight -> ~1 L2 round trip for the whole 32 KB slab)
#pragma unroll
    for (int j = 0; j < 8; ++j) {
      const int idx = (j * 256 + tid) * 8;   // 8 bf16 per v8us
      *(v8us*)&h_lds[idx] = *(const v8us*)&hcur[idx];
    }
    __syncthreads();

    // -------- phase 1: gate slab = h_prev @ Whh^T, A-fragments from LDS
    {
      v8f acc0 = {}, acc1 = {};
      const unsigned short* pa = &h_lds[(mt * 16 + r) * CH + 8 * h];
#pragma unroll
      for (int kk = 0; kk < KT; ++kk) {
        FragU fa;
        fa.us[0] = *(const v8us*)(pa + kk * 32);
        fa.us[1] = *(const v8us*)(pa + kk * 32 + 16);
        acc0 = __builtin_amdgcn_wmma_f32_16x16x32_bf16(false, fa.bf, false, wb0[kk],
                                                       (short)0, acc0, false, false);
        acc1 = __builtin_amdgcn_wmma_f32_16x16x32_bf16(false, fa.bf, false, wb1[kk],
                                                       (short)0, acc1, false, false);
      }
      const int lcol = group * 32 + r;   // slab column for tile pair
#pragma unroll
      for (int i = 0; i < 8; ++i) {
        g_lds[mt * 16 + i + 8 * h][lcol]      = acc0[i];
        g_lds[mt * 16 + i + 8 * h][lcol + 16] = acc1[i];
      }
    }
    __syncthreads();

    // -------- phase 2: LSTM cell for 1024 (b,lh) elements, 4 per thread
#pragma unroll
    for (int k2 = 0; k2 < 4; ++k2) {
      const int idx = k2 * 256 + tid;
      const int b = idx >> 5;
      const int lh = idx & 31;
      const size_t xbase = ((size_t)b * CT + t) * CG + hbase + lh;
      float gi = g_lds[b][0 * HS + lh] + xg[xbase + 0 * CH];
      float gf = g_lds[b][1 * HS + lh] + xg[xbase + 1 * CH];
      float gg = g_lds[b][2 * HS + lh] + xg[xbase + 2 * CH];
      float go = g_lds[b][3 * HS + lh] + xg[xbase + 3 * CH];
      float c = sigf(gf) * creg[k2] + sigf(gi) * tanhf(gg);
      float hv = sigf(go) * tanhf(c);
      creg[k2] = c;
      unsigned short hb16 = f2bf(hv);
      hcur[(size_t)b * CH + hbase + lh] = hb16;
      hcat[((size_t)b * CT + t) * (2 * CH) + (size_t)dir * CH + hbase + lh] = hb16;
    }

    grid_barrier(ctr, gen, LSTM_BLOCKS);
  }
}

// ---------------------------------------------------------------------------
// CRF: golden score + forward (logsumexp) recursion, one block per batch elem.
__global__ __launch_bounds__(64) void crf_kernel(
    const float* __restrict__ emit, const float* __restrict__ trans,
    const int* __restrict__ targets, float* __restrict__ part) {
  const int b = blockIdx.x;
  const int tid = threadIdx.x;
  __shared__ float tr[CL * CL];
  __shared__ float alpha[2][CL];
  __shared__ float red[64];

  for (int i = tid; i < CL * CL; i += 64) tr[i] = trans[i];
  __syncthreads();

  // golden path score (deterministic fixed-order reduction)
  float gsum = 0.f;
  for (int t = tid; t < CT; t += 64) {
    int tgt  = targets[b * CT + t];
    int prev = (t == 0) ? BOS : targets[b * CT + t - 1];
    gsum += emit[((size_t)b * CT + t) * CL + tgt] + tr[prev * CL + tgt];
  }
  red[tid] = gsum;
  __syncthreads();
  for (int s = 32; s > 0; s >>= 1) {
    if (tid < s) red[tid] += red[tid + s];
    __syncthreads();
  }
  const float golden = red[0];

  // forward recursion
  if (tid < CL) alpha[0][tid] = emit[((size_t)b * CT) * CL + tid] + tr[BOS * CL + tid];
  __syncthreads();
  int cur = 0;
  for (int t = 1; t < CT; ++t) {
    if (tid < CL) {
      const int j = tid;
      float m = -3.4e38f;
      for (int i = 0; i < CL; ++i) m = fmaxf(m, alpha[cur][i] + tr[i * CL + j]);
      float s = 0.f;
      for (int i = 0; i < CL; ++i) s += expf(alpha[cur][i] + tr[i * CL + j] - m);
      alpha[cur ^ 1][j] = m + logf(s) + emit[((size_t)b * CT + t) * CL + j];
    }
    __syncthreads();
    cur ^= 1;
  }
  if (tid == 0) part[b] = alpha[cur][EOS] - golden;
}

__global__ void final_kernel(const float* __restrict__ part, float* __restrict__ out) {
  if (blockIdx.x == 0 && threadIdx.x == 0) {
    float s = 0.f;
    for (int b = 0; b < CB; ++b) s += part[b];   // fixed order: deterministic
    out[0] = s / (float)CB;
  }
}

// ---------------------------------------------------------------------------
extern "C" void kernel_launch(void* const* d_in, const int* in_sizes, int n_in,
                              void* d_out, int out_size, void* d_ws, size_t ws_size,
                              hipStream_t stream) {
  (void)in_sizes; (void)n_in; (void)out_size; (void)ws_size;
  // inputs (setup_inputs order)
  const int*   src    = (const int*)  d_in[0];
  /* lengths d_in[1] unused (all full T) */
  const int*   tgts   = (const int*)  d_in[2];
  const float* emb    = (const float*)d_in[3];
  const float* wih_f  = (const float*)d_in[4];
  const float* whh_f  = (const float*)d_in[5];
  const float* b_f    = (const float*)d_in[6];
  const float* wih_b  = (const float*)d_in[7];
  const float* whh_b  = (const float*)d_in[8];
  const float* b_b    = (const float*)d_in[9];
  const float* w_emit = (const float*)d_in[10];
  const float* b_emit = (const float*)d_in[11];
  const float* trans  = (const float*)d_in[12];
  float* out = (float*)d_out;

  // workspace carve-up (256B aligned)
  size_t off = 0;
  auto take = [&](size_t bytes) -> void* {
    off = (off + 255) & ~(size_t)255;
    void* p = (char*)d_ws + off;
    off += bytes;
    return p;
  };
  unsigned*       hdr     = (unsigned*)      take(256);                      // barrier ctr+gen
  float*          part    = (float*)         take(CB * sizeof(float));
  unsigned short* wihf_bf = (unsigned short*)take((size_t)CG * CE * 2);
  unsigned short* wihb_bf = (unsigned short*)take((size_t)CG * CE * 2);
  unsigned short* whhf_bf = (unsigned short*)take((size_t)CG * CH * 2);
  unsigned short* whhb_bf = (unsigned short*)take((size_t)CG * CH * 2);
  unsigned short* wem_bf  = (unsigned short*)take((size_t)CL * 2 * CH * 2);
  unsigned short* x_bf    = (unsigned short*)take((size_t)CBT * CE * 2);
  float*          xg_f    = (float*)         take((size_t)CBT * CG * 4);
  float*          xg_b    = (float*)         take((size_t)CBT * CG * 4);
  unsigned short* hcat    = (unsigned short*)take((size_t)CBT * 2 * CH * 2);
  unsigned short* hcur_f  = (unsigned short*)take((size_t)CB * CH * 2);
  unsigned short* hcur_b  = (unsigned short*)take((size_t)CB * CH * 2);
  float*          emit    = (float*)         take((size_t)CBT * CL * 4);

  // zero barrier state and initial h (graph-capture-safe async memsets)
  hipMemsetAsync(hdr, 0, 256, stream);
  hipMemsetAsync(hcur_f, 0, (size_t)CB * CH * 2, stream);
  hipMemsetAsync(hcur_b, 0, (size_t)CB * CH * 2, stream);

  // weights -> bf16
  auto cvt = [&](const float* s, unsigned short* d, int n) {
    f2bf_kernel<<<(n + 255) / 256, 256, 0, stream>>>(s, d, n);
  };
  cvt(wih_f,  wihf_bf, CG * CE);
  cvt(wih_b,  wihb_bf, CG * CE);
  cvt(whh_f,  whhf_bf, CG * CH);
  cvt(whh_b,  whhb_bf, CG * CH);
  cvt(w_emit, wem_bf,  CL * 2 * CH);

  // embedding gather -> bf16
  embed_kernel<<<CBT, CE, 0, stream>>>(src, emb, x_bf);

  // input projections: xg = x @ Wih^T + b   [16384,256]x[256,2048], 16x64/wave
  {
    dim3 grid(CBT / 16, CG / 64);
    wmma_gemm_bias<4><<<grid, 32, 0, stream>>>(x_bf, CE, wihf_bf, CE, b_f, xg_f, CG, CE);
    wmma_gemm_bias<4><<<grid, 32, 0, stream>>>(x_bf, CE, wihb_bf, CE, b_b, xg_b, CG, CE);
  }

  // bidirectional LSTM recurrence (persistent, grid-barrier per step)
  {
    dim3 grid(NBH, 2);
    lstm_kernel<<<grid, 256, 0, stream>>>(xg_f, xg_b, whhf_bf, whhb_bf,
                                          hcur_f, hcur_b, hcat, hdr);
  }

  // emit = hcat @ W_emit^T + b_emit   [16384,1024]x[1024,48], 16x48/wave
  {
    dim3 grid(CBT / 16, 1);
    wmma_gemm_bias<3><<<grid, 32, 0, stream>>>(hcat, 2 * CH, wem_bf, 2 * CH, b_emit,
                                               emit, CL, 2 * CH);
  }

  // CRF per batch element, then deterministic final combine
  crf_kernel<<<CB, 64, 0, stream>>>(emit, trans, tgts, part);
  final_kernel<<<1, 32, 0, stream>>>(part, out);
}
